// DotProductAttention_11038065951419
// MI455X (gfx1250) — matrix operand order repaired and checked by reading, not compiled, not run on hardware
//
#include <hip/hip_runtime.h>
#include <hip/hip_bf16.h>
#include <math.h>

// ---------------------------------------------------------------------------
// Flash-attention, B=32, Q=K=2048, D=128 (fp32 in/out), query-row masking.
// v_wmma_f32_16x16x32_f16 path, wave32. Pre-pass converts K->f16 and V->f16
// transposed into d_ws so the hot loop does no fp32->f16 conversion.
// Main kernel: block = 128 threads (4 waves); wave tile = 32 q rows;
// k-tile BC = 32. BOTH K and V^T tiles live in LDS, double-buffered and
// staged one tile ahead, so hot-loop fragment reads are short-latency
// ds_loads instead of global loads (the scheduler refuses to keep global
// loads in flight in registers at this VGPR pressure).
// ---------------------------------------------------------------------------

typedef __attribute__((ext_vector_type(16))) _Float16 v16h;
typedef __attribute__((ext_vector_type(8)))  _Float16 v8h;
typedef __attribute__((ext_vector_type(4)))  _Float16 v4h;
typedef __attribute__((ext_vector_type(8)))  float    v8f;

#define BATCH 32
#define QLEN  2048
#define KLEN  2048
#define DHEAD 128
#define BC    32       // k rows per iteration (2 x 16-wide column tiles)
#define NITER (KLEN / BC)
#define KT_STRIDE 136  // halves per K row in LDS (272B: 16B-aligned, bank-spread)
#define VT_STRIDE 40   // halves per V^T row in LDS (80B)
#define FB_BC 32       // fallback k-tile
#define FB_VT_STRIDE 40

static __device__ __forceinline__ v8f wmma_f16(v16h a, v16h b, v8f c) {
    return __builtin_amdgcn_wmma_f32_16x16x32_f16(
        false, a, false, b, (short)0, c, false, false);
}

// ---- Pre-pass 1: straight fp32 -> f16 conversion (for K) ------------------
__global__ __launch_bounds__(256)
void cvt_f16_kernel(const float* __restrict__ in, _Float16* __restrict__ out) {
    const size_t i = ((size_t)blockIdx.x * 256 + threadIdx.x) * 4;
    const float4 f = *(const float4*)(in + i);
    v4h h;
    h[0] = (_Float16)f.x; h[1] = (_Float16)f.y;
    h[2] = (_Float16)f.z; h[3] = (_Float16)f.w;
    *(v4h*)(out + i) = h;
}

// ---- Pre-pass 2: V[b][k][d] fp32 -> Vt[b][d][k] f16 (LDS-tiled) -----------
__global__ __launch_bounds__(256)
void transpose_cvt_v_kernel(const float* __restrict__ V, _Float16* __restrict__ Vt) {
    __shared__ float tile[32][33];
    const int b  = blockIdx.z;
    const int k0 = blockIdx.x * 32;
    const int d0 = blockIdx.y * 32;
    const int tx = threadIdx.x & 31;
    const int ty = threadIdx.x >> 5;    // 0..7
    #pragma unroll
    for (int i = 0; i < 4; ++i) {
        const int kl = ty + i * 8;
        tile[kl][tx] = V[((size_t)b * KLEN + k0 + kl) * DHEAD + d0 + tx];
    }
    __syncthreads();
    #pragma unroll
    for (int i = 0; i < 4; ++i) {
        const int dl = ty + i * 8;
        Vt[((size_t)b * DHEAD + d0 + dl) * KLEN + k0 + tx] = (_Float16)tile[tx][dl];
    }
}

// ---- Main kernel: f16 K / transposed f16 V from workspace -----------------
__global__ __launch_bounds__(128)
void flash_attn_wmma_f16kv(const float*    __restrict__ Qm,
                           const _Float16* __restrict__ Kh,
                           const _Float16* __restrict__ Vth,
                           const int*      __restrict__ valid_len,
                           float*          __restrict__ Om) {
    __shared__ _Float16 kt[2][BC * KT_STRIDE];     // K tiles (row-major), 17 KB
    __shared__ _Float16 vt[2][DHEAD * VT_STRIDE];  // V^T tiles, 20 KB
    __shared__ _Float16 pbuf[4][32 * BC];          // per-wave P (32q x 32k), 8 KB

    const int b    = blockIdx.y;
    const int q0   = blockIdx.x * 128;             // 128 q rows per block
    const int tid  = threadIdx.x;
    const int wid  = tid >> 5;
    const int lane = tid & 31;
    const int lo   = lane & 15;
    const int hi   = lane >> 4;

    const int vlen    = valid_len[b];
    const int abase   = hi ? 8 : 0;    // A-layout half-wave offset
    const int cbase16 = hi ? 16 : 0;   // unused; B fragments also use abase

    // ---- Q fragments (two 16-row subtiles), pre-scaled by 1/sqrt(D)
    const float qscale = 0.08838834764831845f;
    v16h aq[2][4];
    #pragma unroll
    for (int u = 0; u < 2; ++u) {
        const int qrow = q0 + wid * 32 + u * 16 + lo;
        const float* qp = Qm + ((size_t)b * QLEN + qrow) * DHEAD;
        #pragma unroll
        for (int c = 0; c < 4; ++c) {
            const float* p0 = qp + c * 32 + abase;
            v16h a;
            #pragma unroll
            for (int i = 0; i < 8; ++i) a[i]     = (_Float16)(p0[i]      * qscale);
            #pragma unroll
            for (int i = 0; i < 8; ++i) a[8 + i] = (_Float16)(p0[16 + i] * qscale);
            aq[u][c] = a;
        }
    }

    v8f  oacc[2][8];
    float mrow[2][8], lrow[2][8];
    #pragma unroll
    for (int u = 0; u < 2; ++u) {
        #pragma unroll
        for (int dt = 0; dt < 8; ++dt)
            #pragma unroll
            for (int r = 0; r < 8; ++r) oacc[u][dt][r] = 0.0f;
        #pragma unroll
        for (int r = 0; r < 8; ++r) { mrow[u][r] = -INFINITY; lrow[u][r] = 0.0f; }
    }

    // ---- Stage K tile (32x128) and V^T tile (128x32) for iteration kj
    auto stageKV = [&](int kj, int bufIdx) {
        // V^T: thread d copies 32 halves (64B) of row d
        {
            const int d = tid;   // 0..127
            const v8h* src = (const v8h*)(Vth + ((size_t)b * DHEAD + d) * KLEN + kj);
            v8h* dst = (v8h*)(&vt[bufIdx][d * VT_STRIDE]);
            #pragma unroll
            for (int i = 0; i < 4; ++i) dst[i] = src[i];
        }
        // K: 4 threads per row, each copies 32 halves (64B)
        {
            const int kv = tid >> 2;            // 0..31 key row
            const int c0 = (tid & 3) * 32;      // half-offset within row
            const v8h* src = (const v8h*)(Kh + ((size_t)b * KLEN + kj + kv) * DHEAD + c0);
            v8h* dst = (v8h*)(&kt[bufIdx][kv * KT_STRIDE + c0]);
            #pragma unroll
            for (int i = 0; i < 4; ++i) dst[i] = src[i];
        }
    };

    stageKV(0, 0);

    for (int j = 0; j < NITER; ++j) {
        const int kj = j * BC;
        __syncthreads();                        // tiles[j&1] staged; other free
        if (j + 1 < NITER) stageKV(kj + BC, (j + 1) & 1);
        const _Float16* ktile = &kt[j & 1][0];
        const _Float16* vtile = &vt[j & 1][0];

        // Prefetch the K/V tiles after next (global_prefetch_b8)
        if (kj + 2 * BC < KLEN) {
            __builtin_prefetch(Kh + ((size_t)b * KLEN + kj + 2 * BC + lane) * DHEAD, 0, 1);
            __builtin_prefetch(Vth + ((size_t)b * DHEAD + lane * 4) * KLEN + kj + 2 * BC, 0, 1);
        }

        // ---- Scores from LDS: c-major, K fragments double-buffered (ds loads)
        v8f s[2][2];
        #pragma unroll
        for (int u = 0; u < 2; ++u)
            #pragma unroll
            for (int t = 0; t < 2; ++t)
                #pragma unroll
                for (int r = 0; r < 8; ++r) s[u][t][r] = 0.0f;

        auto loadKF = [&](int c, v16h bf[2]) {
            #pragma unroll
            for (int t = 0; t < 2; ++t) {
                const v8h* kr = (const v8h*)(ktile + (t * 16 + lo) * KT_STRIDE
                                             + c * 32 + abase);
                const v8h k0 = kr[0];
                const v8h k1 = kr[2];
                v16h f;
                #pragma unroll
                for (int i = 0; i < 8; ++i) { f[i] = k0[i]; f[8 + i] = k1[i]; }
                bf[t] = f;
            }
        };
        auto mmC = [&](const v16h bf[2], int c) {
            #pragma unroll
            for (int t = 0; t < 2; ++t)
                #pragma unroll
                for (int u = 0; u < 2; ++u)
                    s[u][t] = wmma_f16(aq[u][c], bf[t], s[u][t]);
        };

        v16h bfA[2], bfB[2];
        loadKF(0, bfA);
        loadKF(1, bfB);
        mmC(bfA, 0); loadKF(2, bfA);
        mmC(bfB, 1); loadKF(3, bfB);
        mmC(bfA, 2);
        mmC(bfB, 3);

        // ---- Online softmax (16 rows per lane-half)
        #pragma unroll
        for (int u = 0; u < 2; ++u) {
            #pragma unroll
            for (int r = 0; r < 8; ++r) {
                const int qg = q0 + wid * 32 + u * 16 + r + 8 * hi;
                float sv0 = s[u][0][r];
                float sv1 = s[u][1][r];
                if (qg >= vlen) { sv0 = -100000.0f; sv1 = -100000.0f; }

                float mx = fmaxf(sv0, sv1);
                #pragma unroll
                for (int d = 1; d < 16; d <<= 1)
                    mx = fmaxf(mx, __shfl_xor(mx, d));

                const float mnew  = fmaxf(mrow[u][r], mx);
                const float alpha = __expf(mrow[u][r] - mnew);
                const float p0v   = __expf(sv0 - mnew);
                const float p1v   = __expf(sv1 - mnew);

                float rs = p0v + p1v;
                #pragma unroll
                for (int d = 1; d < 16; d <<= 1)
                    rs += __shfl_xor(rs, d);

                lrow[u][r] = lrow[u][r] * alpha + rs;
                mrow[u][r] = mnew;
                #pragma unroll
                for (int dt = 0; dt < 8; ++dt) oacc[u][dt][r] *= alpha;

                const int prow = u * 16 + r + 8 * hi;
                pbuf[wid][prow * BC + lo]      = (_Float16)p0v;
                pbuf[wid][prow * BC + 16 + lo] = (_Float16)p1v;
            }
        }

        // Wave-local LDS RAW between P writes and A-fragment reads
        asm volatile("s_wait_dscnt 0" ::: "memory");

        // ---- P: C-layout -> A-layout (one K=32 chunk per row subtile)
        v16h ap[2];
        #pragma unroll
        for (int u = 0; u < 2; ++u) {
            const v8h* pr = (const v8h*)(pbuf[wid] + (u * 16 + lo) * BC);
            const v8h c0 = pr[hi];
            const v8h c1 = pr[2 + hi];
            v16h a;
            #pragma unroll
            for (int i = 0; i < 8; ++i) { a[i] = c0[i]; a[8 + i] = c1[i]; }
            ap[u] = a;
        }

        // ---- O += P . Vtile ; LDS B-fragments pipelined one step ahead
        auto loadVF = [&](int dt, v16h& bv) {
            const v8h* vr = (const v8h*)(vtile + (dt * 16 + lo) * VT_STRIDE + abase);
            const v8h b0 = vr[0];
            const v8h b1 = vr[2];
            #pragma unroll
            for (int i = 0; i < 8; ++i) { bv[i] = b0[i]; bv[8 + i] = b1[i]; }
        };

        v16h bv[2];
        loadVF(0, bv[0]);
        #pragma unroll
        for (int dt = 0; dt < 8; ++dt) {
            if (dt + 1 < 8) loadVF(dt + 1, bv[(dt + 1) & 1]);
            #pragma unroll
            for (int u = 0; u < 2; ++u)
                oacc[u][dt] = wmma_f16(ap[u], bv[dt & 1], oacc[u][dt]);
        }
    }

    // ---- Epilogue
    #pragma unroll
    for (int u = 0; u < 2; ++u) {
        #pragma unroll
        for (int r = 0; r < 8; ++r) {
            const int qg = q0 + wid * 32 + u * 16 + r + 8 * hi;
            const float inv_l = 1.0f / lrow[u][r];
            float* op = Om + ((size_t)b * QLEN + qg) * DHEAD + lo;
            #pragma unroll
            for (int dt = 0; dt < 8; ++dt)
                op[dt * 16] = oacc[u][dt][r] * inv_l;
        }
    }
}

// ---- Fallback (fp32 K/V converted in-loop, used when d_ws is too small) ---
__global__ __launch_bounds__(128)
void flash_attn_wmma_fallback(const float* __restrict__ Qm,
                              const float* __restrict__ Km,
                              const float* __restrict__ Vm,
                              const int*   __restrict__ valid_len,
                              float*       __restrict__ Om) {
    __shared__ _Float16 vt[DHEAD * FB_VT_STRIDE];
    __shared__ _Float16 pbuf[4][16 * FB_BC];

    const int b    = blockIdx.y;
    const int q0   = blockIdx.x * 64;
    const int tid  = threadIdx.x;
    const int wid  = tid >> 5;
    const int lane = tid & 31;
    const int lo   = lane & 15;
    const int hi   = lane >> 4;

    const int vlen = valid_len[b];
    const int qrow = q0 + wid * 16 + lo;
    const float qscale = 0.08838834764831845f;
    const float* qp = Qm + ((size_t)b * QLEN + qrow) * DHEAD;
    const int abase   = hi ? 8 : 0;
    const int cbase16 = hi ? 16 : 0;

    v16h aq[4];
    #pragma unroll
    for (int c = 0; c < 4; ++c) {
        const float* p0 = qp + c * 32 + abase;
        v16h a;
        #pragma unroll
        for (int i = 0; i < 8; ++i) a[i]     = (_Float16)(p0[i]      * qscale);
        #pragma unroll
        for (int i = 0; i < 8; ++i) a[8 + i] = (_Float16)(p0[16 + i] * qscale);
        aq[c] = a;
    }

    v8f  oacc[8];
    float mrow[8], lrow[8];
    #pragma unroll
    for (int dt = 0; dt < 8; ++dt)
        #pragma unroll
        for (int r = 0; r < 8; ++r) oacc[dt][r] = 0.0f;
    #pragma unroll
    for (int r = 0; r < 8; ++r) { mrow[r] = -INFINITY; lrow[r] = 0.0f; }

    for (int kj = 0; kj < KLEN; kj += FB_BC) {
        __syncthreads();
        {
            const int kv = tid >> 2;
            const int c0 = (tid & 3) * 32;
            const float* vp = Vm + ((size_t)b * KLEN + (kj + kv)) * DHEAD + c0;
            #pragma unroll
            for (int i = 0; i < 32; ++i)
                vt[(c0 + i) * FB_VT_STRIDE + kv] = (_Float16)vp[i];
        }
        __syncthreads();

        v8f s0, s1;
        #pragma unroll
        for (int r = 0; r < 8; ++r) { s0[r] = 0.0f; s1[r] = 0.0f; }
        #pragma unroll
        for (int t = 0; t < 2; ++t) {
            const int krow = kj + t * 16 + lo;
            const float* kp = Km + ((size_t)b * KLEN + krow) * DHEAD;
            v8f s = t ? s1 : s0;
            #pragma unroll
            for (int c = 0; c < 4; ++c) {
                const float* p0 = kp + c * 32 + abase;
                v16h bf;
                #pragma unroll
                for (int i = 0; i < 16; ++i) bf[i] = (_Float16)p0[i + (i & 8)];
                s = wmma_f16(aq[c], bf, s);
            }
            if (t) s1 = s; else s0 = s;
        }

        #pragma unroll
        for (int r = 0; r < 8; ++r) {
            const int qg = q0 + wid * 16 + r + 8 * hi;
            float sv0 = s0[r];
            float sv1 = s1[r];
            if (qg >= vlen) { sv0 = -100000.0f; sv1 = -100000.0f; }

            float mx = fmaxf(sv0, sv1);
            #pragma unroll
            for (int d = 1; d < 16; d <<= 1)
                mx = fmaxf(mx, __shfl_xor(mx, d));

            const float mnew  = fmaxf(mrow[r], mx);
            const float alpha = __expf(mrow[r] - mnew);
            const float p0v   = __expf(sv0 - mnew);
            const float p1v   = __expf(sv1 - mnew);

            float rs = p0v + p1v;
            #pragma unroll
            for (int d = 1; d < 16; d <<= 1)
                rs += __shfl_xor(rs, d);

            lrow[r] = lrow[r] * alpha + rs;
            mrow[r] = mnew;
            #pragma unroll
            for (int dt = 0; dt < 8; ++dt) oacc[dt][r] *= alpha;

            const int prow = r + 8 * hi;
            pbuf[wid][prow * FB_BC + lo]      = (_Float16)p0v;
            pbuf[wid][prow * FB_BC + 16 + lo] = (_Float16)p1v;
        }

        asm volatile("s_wait_dscnt 0" ::: "memory");

        v16h ap;
        {
            const _Float16* pr = &pbuf[wid][lo * FB_BC];
            #pragma unroll
            for (int i = 0; i < 8; ++i) ap[i]     = pr[abase + i];
            #pragma unroll
            for (int i = 0; i < 8; ++i) ap[8 + i] = pr[abase + 16 + i];
        }

        #pragma unroll
        for (int dt = 0; dt < 8; ++dt) {
            const _Float16* vr = &vt[(dt * 16 + lo) * FB_VT_STRIDE + abase];
            v16h bv;
            #pragma unroll
            for (int i = 0; i < 16; ++i) bv[i] = vr[i + (i & 8)];
            oacc[dt] = wmma_f16(ap, bv, oacc[dt]);
        }
    }

    #pragma unroll
    for (int r = 0; r < 8; ++r) {
        const int qg = q0 + wid * 16 + r + 8 * hi;
        const float inv_l = 1.0f / lrow[r];
        float* op = Om + ((size_t)b * QLEN + qg) * DHEAD + lo;
        #pragma unroll
        for (int dt = 0; dt < 8; ++dt)
            op[dt * 16] = oacc[dt][r] * inv_l;
    }
}

extern "C" void kernel_launch(void* const* d_in, const int* in_sizes, int n_in,
                              void* d_out, int out_size, void* d_ws, size_t ws_size,
                              hipStream_t stream) {
    const float* q  = (const float*)d_in[0];
    const float* k  = (const float*)d_in[1];
    const float* v  = (const float*)d_in[2];
    const int*   vl = (const int*)d_in[3];
    float* out = (float*)d_out;

    const size_t nKV  = (size_t)BATCH * KLEN * DHEAD;        // 8.4M elements
    const size_t need = 2 * nKV * sizeof(_Float16);           // ~33.5 MB

    if (ws_size >= need) {
        _Float16* Kh  = (_Float16*)d_ws;
        _Float16* Vth = Kh + nKV;

        cvt_f16_kernel<<<dim3((unsigned)(nKV / (256 * 4))), dim3(256), 0, stream>>>(k, Kh);
        transpose_cvt_v_kernel<<<dim3(KLEN / 32, DHEAD / 32, BATCH), dim3(256), 0, stream>>>(v, Vth);

        dim3 grid(QLEN / 128, BATCH);   // 16 x 32 blocks
        flash_attn_wmma_f16kv<<<grid, dim3(128), 0, stream>>>(q, Kh, Vth, vl, out);
    } else {
        dim3 grid(QLEN / 64, BATCH);    // 32 x 32 blocks
        flash_attn_wmma_fallback<<<grid, dim3(128), 0, stream>>>(q, k, v, vl, out);
    }
}
